// Global_block_40346922779169
// MI455X (gfx1250) — compile-verified, matrix operand order checked by reading, not run on hardware
//
#include <hip/hip_runtime.h>

// ---------------------------------------------------------------------------
// Swin block for MI455X (gfx1250): wave32, WMMA f16 -> f32, vectorized frags.
// ---------------------------------------------------------------------------

#define B_   8
#define C_   96
#define H_   224
#define W_   224
#define WS_  7
#define SS_  3
#define NH_  3
#define HD_  32
#define N_   49          // tokens per window
#define NWIN 1024        // (H/WS)*(W/WS)
#define HW_  (H_*W_)     // 50176
#define CHW_ (C_*HW_)    // 4816896
#define CW_  (C_*W_)     // 21504
#define MLPH 384

typedef __attribute__((ext_vector_type(16))) _Float16 v16h;
typedef __attribute__((ext_vector_type(8)))  _Float16 v8h;
typedef __attribute__((ext_vector_type(8)))  float    v8f;

// ---- WMMA fragment helpers (CDNA5 ISA 7.12.2, wave32) ----------------------
// element e of a 16-half fragment: k = (e>=8?16:0) + khalf*8 + (e&7)
// => e=0..7 and e=8..15 are each 8 contiguous halves: two 16B loads.

__device__ __forceinline__ v16h frag_row(const _Float16* __restrict__ base,
                                         int row, int ld, int k0, int khalf) {
  const _Float16* p = base + row * ld + k0 + (khalf << 3);
  v8h lo = *(const v8h*)(p);
  v8h hi = *(const v8h*)(p + 16);
  return __builtin_shufflevector(lo, hi, 0, 1, 2, 3, 4, 5, 6, 7, 8, 9, 10, 11,
                                 12, 13, 14, 15);
}

__device__ __forceinline__ v8f wmma_f16(v16h a, v16h b, v8f c) {
  return __builtin_amdgcn_wmma_f32_16x16x32_f16(false, a, false, b,
                                                (short)0, c, false, false);
}

// ---- tiny prep kernels -----------------------------------------------------

// transpose KxN f32 -> NxK f16 (so B-fragments are contiguous row loads)
__global__ void cvt_transpose(const float* __restrict__ src,
                              _Float16* __restrict__ dst, int K, int N) {
  int i = blockIdx.x * blockDim.x + threadIdx.x;
  if (i < K * N) {
    int k = i / N, n = i % N;
    dst[n * K + k] = (_Float16)src[k * N + n];
  }
}

__global__ void build_bias_tab(const float* __restrict__ rpb,
                               float* __restrict__ biasT) {
  int i = blockIdx.x * blockDim.x + threadIdx.x;  // [h][n][m]
  if (i >= NH_ * N_ * N_) return;
  int h = i / (N_ * N_);
  int rest = i % (N_ * N_);
  int n = rest / N_, m = rest % N_;
  int i1 = n / WS_, j1 = n % WS_, i2 = m / WS_, j2 = m % WS_;
  int idx = (i1 - i2 + WS_ - 1) * (2 * WS_ - 1) + (j1 - j2 + WS_ - 1);
  biasT[i] = rpb[idx * NH_ + h];
}

// ---- fused window attention ------------------------------------------------
// one block (8 wave32) per window.
// LDS layout (bytes):
//   [0,12288)      xn   f16 [64][96]  normalized input; later attn-out
//   [12288,24576)  qmat f16 [64][96]
//   [24576,36864)  kmat f16 [64][96]
//   [36864,49152)  vT   f16 [96][64]  (V transposed: [dim][token])
//   [49152,57344)  sT   f16 [64][64]  scores transposed: [col][row]
//   [57344,65536)  pbuf f16 [64][64]  softmax probs, row-major
//   stage f32 [49][96] (18816 B) aliases [12288,31104) during phase 1

__global__ __launch_bounds__(256) void swin_attn_kernel(
    const float* __restrict__ x, const float* __restrict__ mask,
    const float* __restrict__ n1g, const float* __restrict__ n1b,
    const _Float16* __restrict__ qkvWtr, const float* __restrict__ qkvB,
    const _Float16* __restrict__ projWtr, const float* __restrict__ projB,
    const float* __restrict__ biasT, float* __restrict__ x1) {
  __shared__ __align__(16) char smem[65536];
  _Float16* xn    = (_Float16*)(smem);
  _Float16* qmat  = (_Float16*)(smem + 12288);
  _Float16* kmat  = (_Float16*)(smem + 24576);
  _Float16* vT    = (_Float16*)(smem + 36864);
  _Float16* sT    = (_Float16*)(smem + 49152);
  _Float16* pbuf  = (_Float16*)(smem + 57344);
  float*    stage = (float*)(smem + 12288);

  const int tid   = threadIdx.x;
  const int lane  = tid & 31;
  const int wv    = tid >> 5;  // wave id 0..7
  const int khalf = lane >> 4;
  const int l15   = lane & 15;

  const int blk = blockIdx.x;
  const int wid = blk & (NWIN - 1);
  const int b   = blk >> 10;
  const int wh  = wid >> 5, ww = wid & 31;

  // ---- phase 1: gather window (with -SS roll), LN1 -> xn f16, zero pad rows
  for (int it = 0; it < 19; ++it) {
    int idx = tid + it * 256;
    if (idx < N_ * C_) {
      int t = idx / C_, c = idx % C_;
      int h = (wh * WS_ + t / WS_ + SS_) % H_;
      int w = (ww * WS_ + t % WS_ + SS_) % W_;
      stage[idx] = x[((b * C_ + c) * H_ + h) * W_ + w];
    }
  }
  __syncthreads();
  // zero xn pad rows (keeps padded q/k/v finite: bias only)
  for (int idx = tid; idx < (64 - N_) * C_; idx += 256)
    xn[N_ * C_ + idx] = (_Float16)0.0f;
  if (tid < N_) {
    float mu = 0.f;
    for (int c = 0; c < C_; ++c) mu += stage[tid * C_ + c];
    mu *= (1.f / C_);
    float m2 = 0.f;
    for (int c = 0; c < C_; ++c) {
      float d = stage[tid * C_ + c] - mu;
      m2 += d * d;
    }
    float rinv = rsqrtf(m2 * (1.f / C_) + 1e-5f);
    for (int c = 0; c < C_; ++c)
      xn[tid * C_ + c] =
          (_Float16)((stage[tid * C_ + c] - mu) * rinv * n1g[c] + n1b[c]);
  }
  __syncthreads();

  // ---- phase 2: QKV = xn @ qkv_w + b  (64x288, 4x18 tiles, K=3x32)
  for (int i = 0; i < 9; ++i) {
    int tile = wv * 9 + i;
    int m0 = (tile / 18) * 16, n0 = (tile % 18) * 16;
    v8f acc = {};
#pragma unroll
    for (int kc = 0; kc < 3; ++kc) {
      v16h a  = frag_row(xn, m0 + l15, C_, kc * 32, khalf);
      v16h bf = frag_row(qkvWtr, n0 + l15, C_, kc * 32, khalf);
      acc = wmma_f16(a, bf, acc);
    }
    int n = n0 + l15;
    float bias = qkvB[n];
    if (n0 >= 2 * C_) {  // V: store transposed, packed 16B
      int c = n - 2 * C_;
      v8h pk;
#pragma unroll
      for (int r = 0; r < 8; ++r) pk[r] = (_Float16)(acc[r] + bias);
      *(v8h*)(vT + c * 64 + m0 + (khalf << 3)) = pk;
    } else {
      float scl = (n0 < C_) ? 0.17677669529663687f : 1.0f;  // q * HD^-0.5
      _Float16* dstm = (n0 < C_) ? qmat : kmat;
      int c = (n0 < C_) ? n : (n - C_);
#pragma unroll
      for (int r = 0; r < 8; ++r) {
        int m = m0 + r + (khalf << 3);
        dstm[m * C_ + c] = (_Float16)((acc[r] + bias) * scl);
      }
    }
  }
  __syncthreads();

  // ---- phase 3: per-head attention
  for (int hd = 0; hd < NH_; ++hd) {
    // scores = q @ k^T (16 tiles, K=32); store transposed f16, packed
    for (int i = 0; i < 2; ++i) {
      int tile = wv * 2 + i;
      int m0 = (tile >> 2) * 16, n0 = (tile & 3) * 16;
      v8f acc = {};
      v16h a  = frag_row(qmat, m0 + l15, C_, hd * HD_, khalf);
      v16h bf = frag_row(kmat, n0 + l15, C_, hd * HD_, khalf);
      acc = wmma_f16(a, bf, acc);
      v8h pk;
#pragma unroll
      for (int r = 0; r < 8; ++r) pk[r] = (_Float16)acc[r];
      *(v8h*)(sT + (n0 + l15) * 64 + m0 + (khalf << 3)) = pk;
    }
    __syncthreads();
    // softmax over valid 49 cols (bias + shifted-window mask)
    if (tid < 64) {
      if (tid < N_) {
        const float* brow = biasT + (hd * N_ + tid) * N_;
        const float* mrow = mask + (wid * N_ + tid) * N_;
        float mx = -1e30f;
        for (int cc = 0; cc < N_; ++cc)
          mx = fmaxf(mx, (float)sT[cc * 64 + tid] + brow[cc] + mrow[cc]);
        float sum = 0.f;
        for (int cc = 0; cc < N_; ++cc)
          sum += __expf((float)sT[cc * 64 + tid] + brow[cc] + mrow[cc] - mx);
        float inv = 1.f / sum;
        for (int cc = 0; cc < N_; ++cc)
          pbuf[tid * 64 + cc] = (_Float16)(
              __expf((float)sT[cc * 64 + tid] + brow[cc] + mrow[cc] - mx) *
              inv);
        for (int cc = N_; cc < 64; ++cc) pbuf[tid * 64 + cc] = (_Float16)0.0f;
      } else {
        for (int cc = 0; cc < 64; ++cc) pbuf[tid * 64 + cc] = (_Float16)0.0f;
      }
    }
    __syncthreads();
    // out_h = P @ V (8 tiles, K=2x32); B from vT [dim][token]
    {
      int m0 = (wv >> 1) * 16, n0 = (wv & 1) * 16;
      v8f acc = {};
#pragma unroll
      for (int kc = 0; kc < 2; ++kc) {
        v16h a  = frag_row(pbuf, m0 + l15, 64, kc * 32, khalf);
        v16h bf = frag_row(vT, hd * HD_ + n0 + l15, 64, kc * 32, khalf);
        acc = wmma_f16(a, bf, acc);
      }
#pragma unroll
      for (int r = 0; r < 8; ++r) {
        int m = m0 + r + (khalf << 3);
        xn[m * C_ + hd * HD_ + n0 + l15] = (_Float16)acc[r];  // attn-out
      }
    }
    __syncthreads();
  }

  // ---- phase 4: proj + window-reverse + roll + bug-faithful reshape + resid
  for (int i = 0; i < 3; ++i) {
    int tile = wv * 3 + i;
    int m0 = (tile / 6) * 16, n0 = (tile % 6) * 16;
    v8f acc = {};
#pragma unroll
    for (int kc = 0; kc < 3; ++kc) {
      v16h a  = frag_row(xn, m0 + l15, C_, kc * 32, khalf);
      v16h bf = frag_row(projWtr, n0 + l15, C_, kc * 32, khalf);
      acc = wmma_f16(a, bf, acc);
    }
    int n = n0 + l15;
    float pb = projB[n];
#pragma unroll
    for (int r = 0; r < 8; ++r) {
      int m = m0 + r + (khalf << 3);
      if (m < N_) {
        int h2 = (wh * WS_ + m / WS_ + SS_) % H_;
        int w2 = (ww * WS_ + m % WS_ + SS_) % W_;
        // (B,H,W,C) -> permute(0,1,3,2) -> raw reshape (B,C,H,W)
        int dst = b * CHW_ + h2 * CW_ + n * W_ + w2;
        x1[dst] = x[dst] + acc[r] + pb;
      }
    }
  }
}

// ---- fused MLP -------------------------------------------------------------
// LDS: [0,12288) a f16 [64][96];  [12288,61440) h f16 [64][384]
//      stage f32 [64][96] aliases the h region during phase 1.

__global__ __launch_bounds__(256) void swin_mlp_kernel(
    const float* __restrict__ x1, const float* __restrict__ n2g,
    const float* __restrict__ n2b, const _Float16* __restrict__ fc1Wtr,
    const float* __restrict__ fc1B, const _Float16* __restrict__ fc2Wtr,
    const float* __restrict__ fc2B, float* __restrict__ outp) {
  __shared__ __align__(16) char smem[61440];
  _Float16* albs  = (_Float16*)smem;            // [64][96]
  _Float16* hlds  = (_Float16*)(smem + 12288);  // [64][384]
  float*    stage = (float*)(smem + 12288);     // [64][96] phase-1 alias

  const int tid   = threadIdx.x;
  const int lane  = tid & 31;
  const int wv    = tid >> 5;
  const int khalf = lane >> 4;
  const int l15   = lane & 15;
  const int rbase = blockIdx.x * 64;

  // phase 1: gather 64 rows (bug-faithful permuted indexing) into stage
  for (int it = 0; it < 24; ++it) {
    int idx = tid + it * 256;  // 64*96 = 6144 exactly
    int rl = idx / C_, j = idx % C_;
    int row_g = rbase + rl;
    int bb = row_g / HW_, rr = row_g % HW_;
    int f = rr * C_ + j;  // index into (C,W,H) flat
    int c = f / HW_, rem = f % HW_;
    int w = rem / H_, hh = rem % H_;
    stage[idx] = x1[bb * CHW_ + c * HW_ + hh * W_ + w];
  }
  __syncthreads();
  // LN2 per row -> f16
  if (tid < 64) {
    float mu = 0.f;
    for (int c = 0; c < C_; ++c) mu += stage[tid * C_ + c];
    mu *= (1.f / C_);
    float m2 = 0.f;
    for (int c = 0; c < C_; ++c) {
      float d = stage[tid * C_ + c] - mu;
      m2 += d * d;
    }
    float rinv = rsqrtf(m2 * (1.f / C_) + 1e-5f);
    for (int c = 0; c < C_; ++c)
      albs[tid * C_ + c] =
          (_Float16)((stage[tid * C_ + c] - mu) * rinv * n2g[c] + n2b[c]);
  }
  __syncthreads();

  // phase 2: h = gelu(a @ fc1_w + b)  (64x384, 4x24 tiles, K=3x32)
  for (int i = 0; i < 12; ++i) {
    int tile = wv + 8 * i;
    int m0 = (tile / 24) * 16, n0 = (tile % 24) * 16;
    v8f acc = {};
#pragma unroll
    for (int kc = 0; kc < 3; ++kc) {
      v16h a  = frag_row(albs, m0 + l15, C_, kc * 32, khalf);
      v16h bf = frag_row(fc1Wtr, n0 + l15, C_, kc * 32, khalf);
      acc = wmma_f16(a, bf, acc);
    }
    float bias = fc1B[n0 + l15];
#pragma unroll
    for (int r = 0; r < 8; ++r) {
      int m = m0 + r + (khalf << 3);
      float v = acc[r] + bias;
      float g = 0.5f * v * (1.f + erff(v * 0.70710678118654752f));  // exact
      hlds[m * MLPH + n0 + l15] = (_Float16)g;
    }
  }
  __syncthreads();

  // phase 3: o = h @ fc2_w + b, residual with x1 (raw flat) -> out
  for (int i = 0; i < 3; ++i) {
    int tile = wv * 3 + i;
    int m0 = (tile / 6) * 16, n0 = (tile % 6) * 16;
    v8f acc = {};
#pragma unroll
    for (int kc = 0; kc < 12; ++kc) {
      v16h a  = frag_row(hlds, m0 + l15, MLPH, kc * 32, khalf);
      v16h bf = frag_row(fc2Wtr, n0 + l15, MLPH, kc * 32, khalf);
      acc = wmma_f16(a, bf, acc);
    }
    int j = n0 + l15;
    float fb = fc2B[j];
#pragma unroll
    for (int r = 0; r < 8; ++r) {
      int m = m0 + r + (khalf << 3);
      int row_g = rbase + m;
      int bb = row_g / HW_, rr = row_g % HW_;
      int dst = bb * CHW_ + rr * C_ + j;  // raw reshape flat index
      outp[dst] = x1[dst] + acc[r] + fb;
    }
  }
}

// ---------------------------------------------------------------------------

extern "C" void kernel_launch(void* const* d_in, const int* in_sizes, int n_in,
                              void* d_out, int out_size, void* d_ws,
                              size_t ws_size, hipStream_t stream) {
  const float* x     = (const float*)d_in[0];
  const float* mask  = (const float*)d_in[1];
  const float* n1g   = (const float*)d_in[2];
  const float* n1b   = (const float*)d_in[3];
  const float* qkvw  = (const float*)d_in[4];
  const float* qkvb  = (const float*)d_in[5];
  const float* rpb   = (const float*)d_in[6];
  const float* projw = (const float*)d_in[7];
  const float* projb = (const float*)d_in[8];
  const float* n2g   = (const float*)d_in[9];
  const float* n2b   = (const float*)d_in[10];
  const float* fc1w  = (const float*)d_in[11];
  const float* fc1b  = (const float*)d_in[12];
  const float* fc2w  = (const float*)d_in[13];
  const float* fc2b  = (const float*)d_in[14];
  float* outp = (float*)d_out;

  // workspace carve (all weights stored TRANSPOSED [N][K] in f16)
  float*    x1      = (float*)d_ws;  // B*C*H*W f32
  _Float16* qkvWtr  = (_Float16*)((char*)d_ws + (size_t)B_ * CHW_ * 4);
  _Float16* projWtr = qkvWtr + 3 * C_ * C_;   // [288][96]
  _Float16* fc1Wtr  = projWtr + C_ * C_;      // [96][96]
  _Float16* fc2Wtr  = fc1Wtr + MLPH * C_;     // [384][96]
  float*    biasT   = (float*)(fc2Wtr + C_ * MLPH);  // [96][384] then bias tab

  cvt_transpose<<<(C_ * 3 * C_ + 255) / 256, 256, 0, stream>>>(qkvw, qkvWtr,
                                                               C_, 3 * C_);
  cvt_transpose<<<(C_ * C_ + 255) / 256, 256, 0, stream>>>(projw, projWtr, C_,
                                                           C_);
  cvt_transpose<<<(C_ * MLPH + 255) / 256, 256, 0, stream>>>(fc1w, fc1Wtr, C_,
                                                             MLPH);
  cvt_transpose<<<(MLPH * C_ + 255) / 256, 256, 0, stream>>>(fc2w, fc2Wtr,
                                                             MLPH, C_);
  build_bias_tab<<<(NH_ * N_ * N_ + 255) / 256, 256, 0, stream>>>(rpb, biasT);

  swin_attn_kernel<<<B_ * NWIN, 256, 0, stream>>>(
      x, mask, n1g, n1b, qkvWtr, qkvb, projWtr, projb, biasT, x1);

  int mlp_blocks = (B_ * HW_) / 64;  // 6272
  swin_mlp_kernel<<<mlp_blocks, 256, 0, stream>>>(x1, n2g, n2b, fc1Wtr, fc1b,
                                                  fc2Wtr, fc2b, outp);
}
